// MultiHeadAttention_22196390985964
// MI455X (gfx1250) — matrix operand order repaired and checked by reading, not compiled, not run on hardware
//
#include <hip/hip_runtime.h>
#include <stdint.h>

typedef __attribute__((ext_vector_type(16))) _Float16 v16h;
typedef __attribute__((ext_vector_type(8)))  float    v8f;

// ---------------- problem constants ----------------
constexpr int MB_  = 2;     // batch
constexpr int MH_  = 16;    // heads
constexpr int MS_  = 2048;  // seq
constexpr int MD_  = 1024;  // d_model
constexpr int MDH_ = 64;    // head dim

// ---------------- tiling constants ----------------
constexpr int BM = 128, BN = 128, BK = 32;
constexpr int LSTR = 24;    // GEMM LDS row stride in dwords

union Frag {
  v16h     h;
  uint32_t u[8];
  uint4    q[2];
};

__device__ inline uint32_t pk2(float a, float b) {
  union { _Float16 h[2]; uint32_t u; } x;
  x.h[0] = (_Float16)a; x.h[1] = (_Float16)b;
  return x.u;
}

__device__ inline v8f wmma16(const Frag& a, const Frag& b, v8f c) {
  return __builtin_amdgcn_wmma_f32_16x16x32_f16(false, a.h, false, b.h,
                                                (short)0, c, false, false);
}

__device__ inline v8f zero8() {
  v8f z;
#pragma unroll
  for (int i = 0; i < 8; ++i) z[i] = 0.0f;
  return z;
}

// CDNA5 async DMA: 16B global -> LDS, tracked by ASYNCcnt.
__device__ inline void async_cp16(uint32_t lds_off, const void* gaddr) {
  asm volatile("global_load_async_to_lds_b128 %0, %1, off"
               :: "v"(lds_off), "v"((uint64_t)(uintptr_t)gaddr)
               : "memory");
}

// Stage 16 consecutive K-elements of one row into 8 packed-f16 dwords.
__device__ inline void stage16(const float* src, uint32_t* d) {
  const float4* s = (const float4*)src;
  float4 f0 = s[0], f1 = s[1], f2 = s[2], f3 = s[3];
  d[0] = pk2(f0.x, f0.y); d[1] = pk2(f0.z, f0.w);
  d[2] = pk2(f1.x, f1.y); d[3] = pk2(f1.z, f1.w);
  d[4] = pk2(f2.x, f2.y); d[5] = pk2(f2.z, f2.w);
  d[6] = pk2(f3.x, f3.y); d[7] = pk2(f3.z, f3.w);
}
__device__ inline void stage16(const _Float16* src, uint32_t* d) {
  const uint4* s = (const uint4*)src;
  uint4 a = s[0], b = s[1];
  d[0] = a.x; d[1] = a.y; d[2] = a.z; d[3] = a.w;
  d[4] = b.x; d[5] = b.y; d[6] = b.z; d[7] = b.w;
}

// ---------------------------------------------------------------------------
// C[M,N] = A[M,K] @ W[N,K]^T + bias[N]
// MODE 0: f32 out, row-major [M,N]          (final projection)
// MODE 1: f16 out at [b,h,s,d]              (Q, K projections)
// MODE 2: f16 out at [b,h,d,s] (transposed) (V projection)
// ---------------------------------------------------------------------------
template <typename AT, int MODE>
__global__ __launch_bounds__(256) void gemm_wmma(
    const AT* __restrict__ A, const float* __restrict__ W,
    const float* __restrict__ bias, void* __restrict__ Out) {
  __shared__ uint32_t Al[BM][LSTR];
  __shared__ uint32_t Wl[BN][LSTR];
  const int K = MD_, N = MD_;

  const int tid  = threadIdx.x;
  const int Nb   = blockIdx.x * BN;
  const int Mb   = blockIdx.y * BM;
  const int lane = tid & 31, half = lane >> 4, l16 = lane & 15;
  const int wid  = tid >> 5, wr = wid >> 2, wc = wid & 3; // 2x4 wave grid
  const int srow = tid >> 1, sseg = tid & 1;              // staging assignment

  v8f acc[4][2];
#pragma unroll
  for (int mt = 0; mt < 4; ++mt)
#pragma unroll
    for (int nt = 0; nt < 2; ++nt) acc[mt][nt] = zero8();

  for (int kb = 0; kb < K; kb += BK) {
    uint32_t ta[8], tw[8];
    stage16(A + (size_t)(Mb + srow) * K + kb + sseg * 16, ta);
    stage16(W + (size_t)(Nb + srow) * K + kb + sseg * 16, tw);
    if (kb + BK < K) {
      __builtin_prefetch(A + (size_t)(Mb + srow) * K + kb + BK, 0, 0);
      __builtin_prefetch(W + (size_t)(Nb + srow) * K + kb + BK, 0, 0);
    }
    __syncthreads();
    *(uint4*)&Al[srow][sseg * 8]     = make_uint4(ta[0], ta[1], ta[2], ta[3]);
    *(uint4*)&Al[srow][sseg * 8 + 4] = make_uint4(ta[4], ta[5], ta[6], ta[7]);
    *(uint4*)&Wl[srow][sseg * 8]     = make_uint4(tw[0], tw[1], tw[2], tw[3]);
    *(uint4*)&Wl[srow][sseg * 8 + 4] = make_uint4(tw[4], tw[5], tw[6], tw[7]);
    __syncthreads();

    Frag af[4], bf[2];
#pragma unroll
    for (int mt = 0; mt < 4; ++mt) {
      const uint32_t* rp = &Al[wr * 64 + mt * 16 + l16][0];
      af[mt].q[0] = *(const uint4*)(rp + half * 4);
      af[mt].q[1] = *(const uint4*)(rp + 8 + half * 4);
    }
#pragma unroll
    for (int nt = 0; nt < 2; ++nt) {
      const uint32_t* rp = &Wl[wc * 32 + nt * 16 + l16][0];
      bf[nt].q[0] = *(const uint4*)(rp + half * 4);
      bf[nt].q[1] = *(const uint4*)(rp + 8 + half * 4);
    }
#pragma unroll
    for (int mt = 0; mt < 4; ++mt)
#pragma unroll
      for (int nt = 0; nt < 2; ++nt)
        acc[mt][nt] = wmma16(af[mt], bf[nt], acc[mt][nt]);
  }

#pragma unroll
  for (int mt = 0; mt < 4; ++mt) {
#pragma unroll
    for (int nt = 0; nt < 2; ++nt) {
      const int col = Nb + wc * 32 + nt * 16 + l16;
      const float bv = bias[col];
#pragma unroll
      for (int v = 0; v < 8; ++v) {
        const int row = Mb + wr * 64 + mt * 16 + v + 8 * half;
        const float val = acc[mt][nt][v] + bv;
        if (MODE == 0) {
          ((float*)Out)[(size_t)row * N + col] = val;
        } else if (MODE == 1) {
          const int b = row >> 11, s = row & (MS_ - 1);
          const int h = col >> 6, d = col & 63;
          ((_Float16*)Out)[(((size_t)(b * MH_ + h) * MS_) + s) * MDH_ + d] =
              (_Float16)val;
        } else {
          const int b = row >> 11, s = row & (MS_ - 1);
          const int h = col >> 6, d = col & 63;
          ((_Float16*)Out)[(((size_t)(b * MH_ + h) * MDH_) + d) * MS_ + s] =
              (_Float16)val;
        }
      }
    }
  }
}

// ---------------------------------------------------------------------------
// Flash attention: grid (S/64, H, B), 128 threads (4 waves x 16 q-rows).
// 64-key blocks; K/V tiles staged to LDS via async DMA, double-buffered.
// Q,K stored [b,h,s,d] f16; V stored [b,h,d,s] f16; output f16 [B,S,D].
// ---------------------------------------------------------------------------
constexpr int KBLK   = 64;
constexpr int TROWB  = 160;              // tile row stride bytes (128 data + 32 pad)
constexpr int TROWW  = 40;               // in dwords
constexpr int TILEB  = 64 * TROWB;       // 10 KB per tile

__global__ __launch_bounds__(128) void attn_fused(
    const _Float16* __restrict__ Q, const _Float16* __restrict__ Kh,
    const _Float16* __restrict__ Vh, const int* __restrict__ mask,
    _Float16* __restrict__ Oatt) {
  __shared__ uint32_t Ktile[2][64][TROWW];
  __shared__ uint32_t Vtile[2][64][TROWW];
  __shared__ uint32_t Pl[4][16][TROWW];   // per-wave P transpose patch

  const int qb = blockIdx.x, h = blockIdx.y, b = blockIdx.z;
  const int tid = threadIdx.x, lane = tid & 31, half = lane >> 4,
            l16 = lane & 15, wid = tid >> 5;

  const _Float16* Qb = Q  + (size_t)(b * MH_ + h) * MS_ * MDH_;
  const _Float16* Kb = Kh + (size_t)(b * MH_ + h) * MS_ * MDH_;
  const _Float16* Vb = Vh + (size_t)(b * MH_ + h) * MDH_ * MS_;
  const int* mb = mask + (size_t)b * MS_;

  const uint32_t kbase = (uint32_t)(uintptr_t)(void*)&Ktile[0][0][0];
  const uint32_t vbase = (uint32_t)(uintptr_t)(void*)&Vtile[0][0][0];
  const int sr = tid >> 1;             // staging row 0..63
  const int so = (tid & 1) * 64;       // byte segment within 128B row

  // Q A-fragments for this wave's 16 rows (resident whole kernel)
  const int qrow = qb * 64 + wid * 16 + l16;
  Frag qa[2];
#pragma unroll
  for (int c = 0; c < 2; ++c) {
    const _Float16* p0 = Qb + (size_t)qrow * MDH_ + c * 32 + half * 8;
    qa[c].q[0] = *(const uint4*)p0;
    qa[c].q[1] = *(const uint4*)(p0 + 16);
  }

  float mrow[8], lrow[8];
  v8f O[4];
#pragma unroll
  for (int v = 0; v < 8; ++v) { mrow[v] = -3.0e38f; lrow[v] = 0.0f; }
#pragma unroll
  for (int f = 0; f < 4; ++f) O[f] = zero8();

  const float SC = 0.125f;  // 1/sqrt(64)
  const int NBLK = MS_ / KBLK;

  // Issue async DMA for one K/V tile pair (8 x 16B per thread).
  auto issue_tiles = [&](int buf, int k0) {
    const uint32_t klds = kbase + buf * TILEB + sr * TROWB + so;
    const uint32_t vlds = vbase + buf * TILEB + sr * TROWB + so;
    const char* kg = (const char*)(Kb + (size_t)(k0 + sr) * MDH_) + so;
    const char* vg = (const char*)(Vb + (size_t)sr * MS_ + k0) + so;
#pragma unroll
    for (int j = 0; j < 4; ++j) {
      async_cp16(klds + j * 16, kg + j * 16);
      async_cp16(vlds + j * 16, vg + j * 16);
    }
  };

  issue_tiles(0, 0);

  for (int kb = 0; kb < NBLK; ++kb) {
    const int k0  = kb * KBLK;
    const int buf = kb & 1;
    const bool has_next = (kb + 1) < NBLK;
    if (has_next) {
      issue_tiles(buf ^ 1, k0 + KBLK);
      asm volatile("s_wait_asynccnt 0x8" ::: "memory");  // current tile done
    } else {
      asm volatile("s_wait_asynccnt 0x0" ::: "memory");
    }
    __syncthreads();  // whole tile visible to all waves

    // ---- scores: 16 q-rows x 64 k-cols (4 n-chunks x 2 d-steps) ----
    v8f sc4[4];
#pragma unroll
    for (int nc = 0; nc < 4; ++nc) {
      sc4[nc] = zero8();
      const uint32_t* kr = &Ktile[buf][nc * 16 + l16][0];
#pragma unroll
      for (int dc = 0; dc < 2; ++dc) {
        Frag kf;
        kf.q[0] = *(const uint4*)(kr + dc * 16 + half * 4);
        kf.q[1] = *(const uint4*)(kr + dc * 16 + 8 + half * 4);
        sc4[nc] = wmma16(qa[dc], kf, sc4[nc]);
      }
    }

    int mk[4];
#pragma unroll
    for (int nc = 0; nc < 4; ++nc) mk[nc] = mb[k0 + nc * 16 + l16];

    // ---- online softmax over the 64-wide block ----
    float pv[4][8], alpha[8];
#pragma unroll
    for (int v = 0; v < 8; ++v) {
      float a0 = mk[0] ? sc4[0][v] * SC : -1.0e9f;
      float a1 = mk[1] ? sc4[1][v] * SC : -1.0e9f;
      float a2 = mk[2] ? sc4[2][v] * SC : -1.0e9f;
      float a3 = mk[3] ? sc4[3][v] * SC : -1.0e9f;
      float r = fmaxf(fmaxf(a0, a1), fmaxf(a2, a3));
#pragma unroll
      for (int off = 8; off >= 1; off >>= 1)
        r = fmaxf(r, __shfl_xor(r, off, 16));
      const float mn = fmaxf(mrow[v], r);
      const float al = __expf(mrow[v] - mn);
      const float e0 = __expf(a0 - mn), e1 = __expf(a1 - mn);
      const float e2 = __expf(a2 - mn), e3 = __expf(a3 - mn);
      float sm = (e0 + e1) + (e2 + e3);
#pragma unroll
      for (int off = 8; off >= 1; off >>= 1) sm += __shfl_xor(sm, off, 16);
      lrow[v] = lrow[v] * al + sm;
      mrow[v] = mn;
      alpha[v] = al;
      pv[0][v] = e0; pv[1][v] = e1; pv[2][v] = e2; pv[3][v] = e3;
    }
#pragma unroll
    for (int f = 0; f < 4; ++f)
#pragma unroll
      for (int v = 0; v < 8; ++v) O[f][v] *= alpha[v];

    // ---- D-layout -> A-layout transpose of P via wave-private LDS ----
    _Float16* pl = (_Float16*)&Pl[wid][0][0];
#pragma unroll
    for (int v = 0; v < 8; ++v) {
      const int row = v + 8 * half;
#pragma unroll
      for (int nc = 0; nc < 4; ++nc)
        pl[row * (TROWW * 2) + nc * 16 + l16] = (_Float16)pv[nc][v];
    }
    asm volatile("s_wait_dscnt 0" ::: "memory");
    Frag pa[2];
    {
      const uint32_t* rp = &Pl[wid][l16][0];
#pragma unroll
      for (int kc = 0; kc < 2; ++kc) {
        pa[kc].q[0] = *(const uint4*)(rp + kc * 16 + half * 4);
        pa[kc].q[1] = *(const uint4*)(rp + kc * 16 + 8 + half * 4);
      }
    }

    // ---- O += P @ V (4 d-chunks x 2 k-steps) ----
#pragma unroll
    for (int f = 0; f < 4; ++f) {
      const uint32_t* vr = &Vtile[buf][f * 16 + l16][0];
#pragma unroll
      for (int kc = 0; kc < 2; ++kc) {
        Frag vf;
        vf.q[0] = *(const uint4*)(vr + kc * 16 + half * 4);
        vf.q[1] = *(const uint4*)(vr + kc * 16 + 8 + half * 4);
        O[f] = wmma16(pa[kc], vf, O[f]);
      }
    }
    __syncthreads();  // tile consumed; safe for next-next DMA to overwrite
  }

#pragma unroll
  for (int v = 0; v < 8; ++v) lrow[v] = 1.0f / lrow[v];
#pragma unroll
  for (int f = 0; f < 4; ++f) {
#pragma unroll
    for (int v = 0; v < 8; ++v) {
      const int sg  = qb * 64 + wid * 16 + v + 8 * half;
      const int col = h * MDH_ + f * 16 + l16;
      Oatt[(size_t)(b * MS_ + sg) * MD_ + col] =
          (_Float16)(O[f][v] * lrow[v]);
    }
  }
}

// ---------------------------------------------------------------------------
extern "C" void kernel_launch(void* const* d_in, const int* in_sizes, int n_in,
                              void* d_out, int out_size, void* d_ws,
                              size_t ws_size, hipStream_t stream) {
  (void)in_sizes; (void)n_in; (void)out_size; (void)ws_size;
  const float* query = (const float*)d_in[0];
  const float* key_t = (const float*)d_in[1];
  const float* value = (const float*)d_in[2];
  const int*   mask  = (const int*)d_in[3];
  const float* Wq = (const float*)d_in[4];
  const float* bq = (const float*)d_in[5];
  const float* Wk = (const float*)d_in[6];
  const float* bk = (const float*)d_in[7];
  const float* Wv = (const float*)d_in[8];
  const float* bv = (const float*)d_in[9];
  const float* Wo = (const float*)d_in[10];
  const float* bo = (const float*)d_in[11];

  const size_t SEG = (size_t)8 << 20;  // 8 MB per f16 buffer
  char* ws = (char*)d_ws;
  _Float16* qf = (_Float16*)(ws + 0 * SEG);  // [b,h,s,d]
  _Float16* kf = (_Float16*)(ws + 1 * SEG);  // [b,h,s,d]
  _Float16* vf = (_Float16*)(ws + 2 * SEG);  // [b,h,d,s]
  _Float16* af = (_Float16*)(ws + 3 * SEG);  // [B*S, D] f16

  dim3 gg(MD_ / BN, (MB_ * MS_) / BM);  // (8, 32)
  gemm_wmma<float, 1><<<gg, 256, 0, stream>>>(query, Wq, bq, qf);
  gemm_wmma<float, 1><<<gg, 256, 0, stream>>>(key_t, Wk, bk, kf);
  gemm_wmma<float, 2><<<gg, 256, 0, stream>>>(value, Wv, bv, vf);

  attn_fused<<<dim3(MS_ / 64, MH_, MB_), 128, 0, stream>>>(qf, kf, vf, mask, af);

  gemm_wmma<_Float16, 0><<<gg, 256, 0, stream>>>(af, Wo, bo, d_out);
}
